// GATPair_37074157699767
// MI455X (gfx1250) — matrix operand order repaired and checked by reading, not compiled, not run on hardware
//
#include <hip/hip_runtime.h>
#include <hip/hip_bf16.h>

#define NN 50000
#define NE 800000
#define NET (NN + NE)          // edges + self loops
#define INDIM 128
#define HID 64
#define HEADS 2
#define HC 128                 // HEADS*HID
#define NPAIRS 8192
#define OUTDIM 8
#define NEG_SLOPE 0.2f

typedef __attribute__((ext_vector_type(16))) __bf16 v16bf;
typedef __attribute__((ext_vector_type(8)))  float  v8f;

#if __has_builtin(__builtin_amdgcn_global_load_async_to_lds_b128) && \
    __has_builtin(__builtin_amdgcn_s_wait_asynccnt)
#define USE_ASYNC_LDS 1
typedef int v4i_g __attribute__((vector_size(16)));                  // matches builtin proto
typedef __attribute__((address_space(3))) v4i_g* lds_v4i_ptr;        // LDS-qualified pointee
#else
#define USE_ASYNC_LDS 0
#endif

__device__ __forceinline__ unsigned short f2bf_bits(float f) {
  unsigned int u = __float_as_uint(f);
  u += 0x7FFFu + ((u >> 16) & 1u);      // round-to-nearest-even
  return (unsigned short)(u >> 16);
}

__global__ void k_zero_f32(float* p, int n) {
  int i = blockIdx.x * blockDim.x + threadIdx.x;
  if (i < n) p[i] = 0.0f;
}

__global__ void k_cvt_bf16(const float* in, unsigned short* out, int n) {
  int i = blockIdx.x * blockDim.x + threadIdx.x;
  if (i < n) out[i] = f2bf_bits(in[i]);
}

// W [K, Nout] f32 -> Wt [Nout, K] bf16 (so B-fragment loads are K-contiguous)
__global__ void k_w_transpose_bf16(const float* W, unsigned short* Wt, int K, int Nout) {
  int i = blockIdx.x * blockDim.x + threadIdx.x;
  if (i >= K * Nout) return;
  int k = i / Nout, n = i - k * Nout;
  Wt[n * K + k] = f2bf_bits(W[k * Nout + n]);
}

// C[M,Nout] = A[M,K](bf16) * Bt[Nout,K](bf16)^T  (+bias, +relu), one 16x16 tile per wave.
// blockDim.x = 32*(Nout/16); gridDim.x = M/16 (M multiple of 16, K multiple of 32, K<=128).
// The block's shared 16xK A-tile is staged into LDS once (async-to-LDS when available),
// so the waves covering the N direction don't re-read A from global.
__global__ void k_gemm_wmma(const unsigned short* __restrict__ Abits,
                            const unsigned short* __restrict__ Btbits,
                            float* __restrict__ C, int M, int K, int Nout,
                            const float* bias, int do_relu) {
  __shared__ __align__(16) unsigned short aTile[16 * 128];   // up to 4KB

  const int lane = threadIdx.x & 31;
  const int wave = threadIdx.x >> 5;
  const int hlf  = lane >> 4;     // which 16-lane half
  const int lrow = lane & 15;
  const int m0 = blockIdx.x * 16;
  const int n0 = wave * 16;

  // ---- stage A tile [16 x K] bf16 into LDS, 16B chunks ----
  const int nChunks = 2 * K;                 // (16*K*2B)/16B
  for (int t = threadIdx.x; t < nChunks; t += blockDim.x) {
    int row  = (t * 8) / K;
    int kcol = (t * 8) % K;
    const unsigned short* gsrc = Abits + (size_t)(m0 + row) * K + kcol;
    unsigned short* ldst = aTile + t * 8;
#if USE_ASYNC_LDS
    __builtin_amdgcn_global_load_async_to_lds_b128((v4i_g*)gsrc, (lds_v4i_ptr)ldst, 0, 0);
#else
    *(uint4*)ldst = *(const uint4*)gsrc;     // load_b128 + ds_store_b128
#endif
  }
#if USE_ASYNC_LDS
  __builtin_amdgcn_s_wait_asynccnt(0);
#endif
  __syncthreads();

  const __bf16* Bt = (const __bf16*)Btbits;
  v8f acc = {};
  for (int k0 = 0; k0 < K; k0 += 32) {
    v16bf a, b;
    // A 16x32 bf16 fragment from LDS: lane holds row M=lrow.
    //   elems 0..7  -> K = k0 + hlf*8 + i          (one 16B ds_load)
    //   elems 8..15 -> K = k0 + 16 + hlf*8 + (i-8) (one 16B ds_load)
    const __bf16* arow = (const __bf16*)(aTile + lrow * K + k0);
    #pragma unroll
    for (int i = 0; i < 8; ++i)  a[i]     = arow[hlf * 8 + i];
    #pragma unroll
    for (int i = 0; i < 8; ++i)  a[8 + i] = arow[16 + hlf * 8 + i];
    // B 32x16 bf16 fragment from global: lane holds column N=lrow; same K order as A.
    const __bf16* bcol = Bt + (size_t)(n0 + lrow) * K + k0;
    #pragma unroll
    for (int i = 0; i < 8; ++i)  b[i]     = bcol[hlf * 8 + i];
    #pragma unroll
    for (int i = 0; i < 8; ++i)  b[8 + i] = bcol[16 + hlf * 8 + i];

    acc = __builtin_amdgcn_wmma_f32_16x16x32_bf16(
        /*neg_a=*/false, a, /*neg_b=*/false, b,
        /*c_mod=*/(short)0, acc, /*reuse_a=*/false, /*reuse_b=*/false);
  }
  // D layout: elem v -> row m0 + hlf*8 + v, col n0 + lrow
  #pragma unroll
  for (int v = 0; v < 8; ++v) {
    int row = m0 + hlf * 8 + v;
    int col = n0 + lrow;
    float val = acc[v];
    if (bias) val += bias[col];
    if (do_relu) val = fmaxf(val, 0.0f);
    C[(size_t)row * Nout + col] = val;
  }
}

// per (node, head): attention logit halves + init segment max / denom
__global__ void k_attn_prep(const float* __restrict__ h, const float* a_s, const float* a_d,
                            float* al_s, float* al_d, float* mbuf, float* den) {
  int i = blockIdx.x * blockDim.x + threadIdx.x;   // i = n*HEADS + head
  if (i >= NN * HEADS) return;
  int head = i & 1;
  int n = i >> 1;
  const float* hp  = h + (size_t)n * HC + head * HID;
  const float* asb = a_s + head * HID;
  const float* adb = a_d + head * HID;
  float ss = 0.f, sd = 0.f;
  for (int c = 0; c < HID; ++c) { float v = hp[c]; ss += v * asb[c]; sd += v * adb[c]; }
  al_s[i] = ss; al_d[i] = sd;
  mbuf[i] = -__builtin_inff();
  den[i]  = 0.f;
}

__device__ __forceinline__ void atomicMaxF(float* addr, float val) {
  if (val >= 0.f) atomicMax((int*)addr, __float_as_int(val));
  else            atomicMin((unsigned int*)addr, __float_as_uint(val));
}

__global__ void k_edge_logits_max(const int* __restrict__ ei,
                                  const float* __restrict__ al_s, const float* __restrict__ al_d,
                                  float* __restrict__ elog, float* mbuf) {
  int e = blockIdx.x * blockDim.x + threadIdx.x;
  if (e >= NET) return;
  int src, dst;
  if (e < NE) { src = ei[e]; dst = ei[NE + e]; } else { src = dst = e - NE; }  // self loops
  #pragma unroll
  for (int h = 0; h < HEADS; ++h) {
    float v = al_s[src * HEADS + h] + al_d[dst * HEADS + h];
    v = (v > 0.f) ? v : NEG_SLOPE * v;               // leaky relu
    elog[(size_t)e * HEADS + h] = v;
    atomicMaxF(&mbuf[dst * HEADS + h], v);
  }
}

__global__ void k_edge_exp_sum(const int* __restrict__ ei, const float* __restrict__ mbuf,
                               float* __restrict__ elog, float* den) {
  int e = blockIdx.x * blockDim.x + threadIdx.x;
  if (e >= NET) return;
  int dst = (e < NE) ? ei[NE + e] : e - NE;
  #pragma unroll
  for (int h = 0; h < HEADS; ++h) {
    float ex = __expf(elog[(size_t)e * HEADS + h] - mbuf[dst * HEADS + h]);
    elog[(size_t)e * HEADS + h] = ex;
    atomicAdd(&den[dst * HEADS + h], ex);
  }
}

// one thread per (edge, channel); both heads handled per thread
__global__ void k_edge_aggregate(const int* __restrict__ ei,
                                 const float* __restrict__ elog, const float* __restrict__ den,
                                 const float* __restrict__ h, float* agg) {
  long i = (long)blockIdx.x * blockDim.x + threadIdx.x;
  if (i >= (long)NET * HID) return;
  int c = (int)(i & (HID - 1));
  int e = (int)(i >> 6);
  int src, dst;
  if (e < NE) { src = ei[e]; dst = ei[NE + e]; } else { src = dst = e - NE; }
  #pragma unroll
  for (int hd = 0; hd < HEADS; ++hd) {
    float alpha = elog[(size_t)e * HEADS + hd] / den[dst * HEADS + hd];
    atomicAdd(&agg[(size_t)dst * HC + hd * HID + c],
              alpha * h[(size_t)src * HC + hd * HID + c]);
  }
}

// mean over heads + bias + relu -> next-layer features [NN, HID]
__global__ void k_finalize(const float* __restrict__ agg, const float* bias, float* feat) {
  int i = blockIdx.x * blockDim.x + threadIdx.x;
  if (i >= NN * HID) return;
  int c = i & (HID - 1);
  int n = i >> 6;
  float v = 0.5f * (agg[(size_t)n * HC + c] + agg[(size_t)n * HC + HID + c]) + bias[c];
  feat[i] = fmaxf(v, 0.f);
}

__global__ void k_gather_concat(const float* __restrict__ fl, const float* __restrict__ fr,
                                const int* li, const int* ri, unsigned short* merged) {
  int i = blockIdx.x * blockDim.x + threadIdx.x;  // p*128 + j
  if (i >= NPAIRS * 2 * HID) return;
  int j = i & 127;
  int p = i >> 7;
  float v = (j < HID) ? fl[(size_t)li[p] * HID + j] : fr[(size_t)ri[p] * HID + (j - HID)];
  merged[i] = f2bf_bits(v);
}

__global__ void k_fc2(const float* __restrict__ h1, const float* __restrict__ w,
                      const float* __restrict__ b, float* out) {
  int i = blockIdx.x * blockDim.x + threadIdx.x;  // p*OUTDIM + o
  if (i >= NPAIRS * OUTDIM) return;
  int o = i & (OUTDIM - 1);
  int p = i >> 3;
  float acc = b[o];
  for (int c = 0; c < HID; ++c) acc += h1[(size_t)p * HID + c] * w[c * OUTDIM + o];
  out[i] = acc;
}

static inline unsigned cdiv(long a, long b) { return (unsigned)((a + b - 1) / b); }

extern "C" void kernel_launch(void* const* d_in, const int* in_sizes, int n_in,
                              void* d_out, int out_size, void* d_ws, size_t ws_size,
                              hipStream_t stream) {
  const float* x_l = (const float*)d_in[0];
  const float* x_r = (const float*)d_in[1];
  const int* ei_l = (const int*)d_in[2];
  const int* ei_r = (const int*)d_in[3];
  const int* l_idx = (const int*)d_in[4];
  const int* r_idx = (const int*)d_in[5];
  // per-layer params: w, a_s, a_d, b at 6 + 4*set
  const float* W[6];  const float* AS[6]; const float* AD[6]; const float* B[6];
  for (int s = 0; s < 6; ++s) {
    W[s]  = (const float*)d_in[6 + 4 * s + 0];
    AS[s] = (const float*)d_in[6 + 4 * s + 1];
    AD[s] = (const float*)d_in[6 + 4 * s + 2];
    B[s]  = (const float*)d_in[6 + 4 * s + 3];
  }
  const float* fc1_w = (const float*)d_in[30];
  const float* fc1_b = (const float*)d_in[31];
  const float* fc2_w = (const float*)d_in[32];
  const float* fc2_b = (const float*)d_in[33];

  // ---- workspace layout ----
  char* ws = (char*)d_ws;
  size_t off = 0;
  auto alloc = [&](size_t bytes) -> void* {
    off = (off + 255) & ~(size_t)255;
    void* p = ws + off;
    off += bytes;
    return p;
  };
  unsigned short* featbf = (unsigned short*)alloc((size_t)NN * INDIM * 2);
  unsigned short* Wt     = (unsigned short*)alloc((size_t)INDIM * HC * 2);
  float* hbuf  = (float*)alloc((size_t)NN * HC * 4);
  float* agg   = (float*)alloc((size_t)NN * HC * 4);
  float* al_s  = (float*)alloc((size_t)NN * HEADS * 4);
  float* al_d  = (float*)alloc((size_t)NN * HEADS * 4);
  float* mbuf  = (float*)alloc((size_t)NN * HEADS * 4);
  float* den   = (float*)alloc((size_t)NN * HEADS * 4);
  float* elog  = (float*)alloc((size_t)NET * HEADS * 4);
  float* featL = (float*)alloc((size_t)NN * HID * 4);
  float* featR = (float*)alloc((size_t)NN * HID * 4);
  unsigned short* mergedbf = (unsigned short*)alloc((size_t)NPAIRS * 2 * HID * 2);
  float* h1    = (float*)alloc((size_t)NPAIRS * HID * 4);
  (void)ws_size; (void)in_sizes; (void)n_in; (void)out_size;

  const int TB = 256;

  auto run_side = [&](const float* x0, const int* ei, const int* layer_sets, float* featOut) {
    const float* cur = x0;
    int Kin = INDIM;
    for (int l = 0; l < 3; ++l) {
      int s = layer_sets[l];
      // bf16 feature + transposed-weight copies for the WMMA GEMM
      k_cvt_bf16<<<cdiv((long)NN * Kin, TB), TB, 0, stream>>>(cur, featbf, NN * Kin);
      k_w_transpose_bf16<<<cdiv((long)Kin * HC, TB), TB, 0, stream>>>(W[s], Wt, Kin, HC);
      // h = x @ W : [NN, HC], 8 waves/block cover the 128-wide output strip
      k_gemm_wmma<<<NN / 16, 256, 0, stream>>>(featbf, Wt, hbuf, NN, Kin, HC, nullptr, 0);
      // attention logits + segment softmax over dst
      k_attn_prep<<<cdiv((long)NN * HEADS, TB), TB, 0, stream>>>(hbuf, AS[s], AD[s],
                                                                 al_s, al_d, mbuf, den);
      k_edge_logits_max<<<cdiv(NET, TB), TB, 0, stream>>>(ei, al_s, al_d, elog, mbuf);
      k_edge_exp_sum<<<cdiv(NET, TB), TB, 0, stream>>>(ei, mbuf, elog, den);
      k_zero_f32<<<cdiv((long)NN * HC, TB), TB, 0, stream>>>(agg, NN * HC);
      k_edge_aggregate<<<cdiv((long)NET * HID, TB), TB, 0, stream>>>(ei, elog, den, hbuf, agg);
      k_finalize<<<cdiv((long)NN * HID, TB), TB, 0, stream>>>(agg, B[s], featOut);
      cur = featOut;
      Kin = HID;
    }
  };

  // left side uses sets {0,1,2}; right side uses {3,4,2} (reference applies conv3_l to right graph)
  const int sets_l[3] = {0, 1, 2};
  const int sets_r[3] = {3, 4, 2};
  run_side(x_l, ei_l, sets_l, featL);
  run_side(x_r, ei_r, sets_r, featR);

  // head: gather/concat -> FC1 (WMMA, bias+relu fused) -> FC2 (scalar)
  k_gather_concat<<<cdiv((long)NPAIRS * 2 * HID, TB), TB, 0, stream>>>(featL, featR,
                                                                      l_idx, r_idx, mergedbf);
  k_w_transpose_bf16<<<cdiv((long)2 * HID * HID, TB), TB, 0, stream>>>(fc1_w, Wt, 2 * HID, HID);
  k_gemm_wmma<<<NPAIRS / 16, 128, 0, stream>>>(mergedbf, Wt, h1, NPAIRS, 2 * HID, HID,
                                               fc1_b, 1);
  k_fc2<<<cdiv((long)NPAIRS * OUTDIM, TB), TB, 0, stream>>>(h1, fc2_w, fc2_b, (float*)d_out);
}
